// SNNasGNNCellModel_6940667150548
// MI455X (gfx1250) — matrix operand order repaired and checked by reading, not compiled, not run on hardware
//
#include <hip/hip_runtime.h>

// Problem constants (from reference)
#define N_NODES 16
#define K_FAN   4
#define BATCH   16384
#define DIM     64
#define IN_DIM  320           // (K_FAN+1)*DIM
#define KSTEPS  10            // IN_DIM / 32
#define NTILES  4             // DIM / 16
#define ALPHA   0.95f
#define BETA    0.90f

typedef __attribute__((ext_vector_type(16))) __bf16       v16bf;
typedef __attribute__((ext_vector_type(2)))  __bf16       v2bf;
typedef __attribute__((ext_vector_type(8)))  float        v8f;
typedef __attribute__((ext_vector_type(8)))  unsigned int v8u;
typedef __attribute__((ext_vector_type(4)))  unsigned int v4u;

// Pack two floats into one dword of bf16x2; backend selects v_cvt_pk_bf16_f32
// if the target has it, otherwise expands inline. Only used in pre-pass kernels.
static __device__ __forceinline__ unsigned int pk2(float lo, float hi) {
  v2bf p;
  p[0] = (__bf16)lo;
  p[1] = (__bf16)hi;
  return __builtin_bit_cast(unsigned int, p);
}

// ---------------------------------------------------------------------------
// Pass 0: elementwise fp32 -> bf16 conversion (row-major layout preserved).
// Each thread: 8 floats in (2 x b128), 8 bf16 out (1 x b128).
// ---------------------------------------------------------------------------
__global__ void __launch_bounds__(256) cvt_bf16_kernel(const float* __restrict__ src,
                                                       unsigned int* __restrict__ dst,
                                                       int n_groups8) {
  const int i = blockIdx.x * blockDim.x + threadIdx.x;
  if (i >= n_groups8) return;
  const float4* s = (const float4*)src + (size_t)2 * i;
  const float4 a = s[0];
  const float4 b = s[1];
  v4u o;
  o[0] = pk2(a.x, a.y);
  o[1] = pk2(a.z, a.w);
  o[2] = pk2(b.x, b.y);
  o[3] = pk2(b.z, b.w);
  *((v4u*)dst + i) = o;
}

// ---------------------------------------------------------------------------
// Pass 1: pack W [N, 320, 64] fp32 into bf16 B-fragments.
// B 32x16 (KxN) fragment packing used here (contiguous K, same as A): lane = column N (l%16);
// lanes 0-15 hold K=0..15 (VGPR r: K=2r,2r+1), lanes 16-31 hold K=16..31.
// One wave produces one (node, kstep, ntile) fragment = 32 lanes x 32 bytes.
// ---------------------------------------------------------------------------
__global__ void __launch_bounds__(256) pack_w_kernel(const float* __restrict__ W,
                                                     unsigned int* __restrict__ wp) {
  const int wave = (blockIdx.x * blockDim.x + threadIdx.x) >> 5;
  const int l = threadIdx.x & 31;
  if (wave >= N_NODES * KSTEPS * NTILES) return;
  const int node = wave / (KSTEPS * NTILES);
  const int rem  = wave % (KSTEPS * NTILES);
  const int t    = rem / NTILES;
  const int nt   = rem % NTILES;
  const int h  = l >> 4;
  const int lm = l & 15;
  const int ncol = nt * 16 + lm;
  // j-th bf16 slot = W[k = t*32 + 16*h + j][ncol]
  const float* w = W + (size_t)node * IN_DIM * DIM
                     + (size_t)(t * 32 + h * 16) * DIM + ncol;
  v8u o;
#pragma unroll
  for (int i = 0; i < 8; ++i)
    o[i] = pk2(w[(size_t)(2 * i) * DIM], w[(size_t)(2 * i + 1) * DIM]);
  *(v8u*)(wp + ((size_t)wave * 32 + l) * 8) = o;
}

// ---------------------------------------------------------------------------
// Pass 2: per-node GEMM (bf16 WMMA, fp32 accum) + LIF update + stores.
// One wave: 32 batch rows x 64 output dims, K=320 in 10 steps of 32.
// A operands come pre-converted from the bf16 workspace: each K-run of 8
// values is one contiguous global_load_b128 -> zero conversion VALU in-loop.
// ---------------------------------------------------------------------------
__global__ void __launch_bounds__(256) snn_gemm_lif_kernel(
    const unsigned short* __restrict__ inputs16,  // [B, D]    bf16
    const unsigned short* __restrict__ sprev16,   // [N, B, D] bf16
    const float* __restrict__ u_in,               // [N, B, D]
    const float* __restrict__ isyn_in,            // [N, B, D]
    const float* __restrict__ bias,               // [N, D]
    const int*   __restrict__ conn,               // [N, K_FAN]
    const unsigned int* __restrict__ wp,
    float* __restrict__ out_all) {
  const int wave = (blockIdx.x * blockDim.x + threadIdx.x) >> 5;
  if (wave >= N_NODES * (BATCH / 32)) return;   // wave-uniform
  const int l  = threadIdx.x & 31;
  const int h  = l >> 4;
  const int lm = l & 15;
  const int node    = wave >> 9;          // 512 row-blocks per node
  const int rowbase = (wave & 511) << 5;  // 32 rows per wave

  // Source pointer per 64-wide input block: 4 gathered predecessors + inputs
  const unsigned short* src[5];
#pragma unroll
  for (int c = 0; c < K_FAN; ++c)
    src[c] = sprev16 + (size_t)conn[node * K_FAN + c] * (BATCH * DIM);
  src[4] = inputs16;

  v8f acc[2][NTILES];
#pragma unroll
  for (int rt = 0; rt < 2; ++rt)
#pragma unroll
    for (int nt = 0; nt < NTILES; ++nt)
      acc[rt][nt] = (v8f){0.f, 0.f, 0.f, 0.f, 0.f, 0.f, 0.f, 0.f};

#pragma unroll
  for (int t = 0; t < KSTEPS; ++t) {
    const unsigned short* s = src[t >> 1];        // constant after unroll
    const int colbase = ((t & 1) << 5) + (h << 4);

    // A fragments: 16x32 bf16; lanes 0-15: K 0..15, lanes 16-31: K 16..31
    v16bf a[2];
#pragma unroll
    for (int rt = 0; rt < 2; ++rt) {
      const unsigned short* p =
          s + (size_t)(rowbase + rt * 16 + lm) * DIM + colbase;
      const v4u r0 = *(const v4u*)p;          // K run0: 8 bf16, 16 bytes
      const v4u r1 = *(const v4u*)(p + 8);    // K run1: 8 bf16, 16 bytes
      v8u ap;
      ap[0] = r0[0]; ap[1] = r0[1]; ap[2] = r0[2]; ap[3] = r0[3];
      ap[4] = r1[0]; ap[5] = r1[1]; ap[6] = r1[2]; ap[7] = r1[3];
      a[rt] = __builtin_bit_cast(v16bf, ap);
    }

#pragma unroll
    for (int nt = 0; nt < NTILES; ++nt) {
      const unsigned int* bp =
          wp + ((size_t)(((node * KSTEPS + t) * NTILES + nt) * 32 + l)) * 8;
      v16bf bfrag = __builtin_bit_cast(v16bf, *(const v8u*)bp);
      acc[0][nt] = __builtin_amdgcn_wmma_f32_16x16x32_bf16(
          false, a[0], false, bfrag, (short)0, acc[0][nt], false, false);
      acc[1][nt] = __builtin_amdgcn_wmma_f32_16x16x32_bf16(
          false, a[1], false, bfrag, (short)0, acc[1][nt], false, false);
    }
  }

  // Output regions (tuple: out, s_new, u_out, i_new — concatenated flat)
  float* out_final = out_all;
  float* s_new = out_all + (size_t)BATCH * DIM;
  float* u_out = s_new + (size_t)N_NODES * BATCH * DIM;
  float* i_new = u_out + (size_t)N_NODES * BATCH * DIM;

  // C/D 16x16 f32 layout: lanes 0-15: N=lane, M=vgpr; lanes 16-31: M=vgpr+8
#pragma unroll
  for (int nt = 0; nt < NTILES; ++nt) {
    const int dcol = nt * 16 + lm;
    const float bv = bias[node * DIM + dcol];
#pragma unroll
    for (int rt = 0; rt < 2; ++rt) {
#pragma unroll
      for (int r = 0; r < 8; ++r) {
        const int brow = rowbase + rt * 16 + r + h * 8;
        const size_t idx = ((size_t)node * BATCH + brow) * DIM + dcol;
        const float x    = acc[rt][nt][r] + bv;
        const float inew = BETA * isyn_in[idx] + x;
        const float unew = ALPHA * u_in[idx] + (1.0f - ALPHA) * inew;
        const float sv   = (unew - 1.0f > 0.0f) ? 1.0f : 0.0f;  // heaviside fwd
        const float uo   = (sv > 0.5f) ? 0.0f : unew;           // refractory reset
        s_new[idx] = sv;
        u_out[idx] = uo;
        i_new[idx] = inew;
        if (node == N_NODES - 1)
          out_final[(size_t)brow * DIM + dcol] = sv;
      }
    }
  }
}

extern "C" void kernel_launch(void* const* d_in, const int* in_sizes, int n_in,
                              void* d_out, int out_size, void* d_ws, size_t ws_size,
                              hipStream_t stream) {
  const float* inputs = (const float*)d_in[0];
  const float* u      = (const float*)d_in[1];
  const float* i_syn  = (const float*)d_in[2];
  const float* s_prev = (const float*)d_in[3];
  const float* W      = (const float*)d_in[4];
  const float* b      = (const float*)d_in[5];
  const int*   conn   = (const int*)d_in[6];
  float* out = (float*)d_out;

  // Workspace layout (requires ~36.3 MB):
  //   [0)                 s_prev bf16: N*B*D  elements (33.5 MB)
  //   [N*B*D)             inputs bf16: B*D    elements ( 2.0 MB)
  //   [(N+1)*B*D)         packed W fragments (640 KB, 32B-aligned)
  unsigned short* sp16 = (unsigned short*)d_ws;
  unsigned short* in16 = sp16 + (size_t)N_NODES * BATCH * DIM;
  unsigned int*   wp   = (unsigned int*)(in16 + (size_t)BATCH * DIM);

  // Pass 0a: s_prev fp32 -> bf16 (16,777,216 elems / 8 per thread)
  {
    const int n8 = (N_NODES * BATCH * DIM) / 8;
    cvt_bf16_kernel<<<n8 / 256, 256, 0, stream>>>(s_prev, (unsigned int*)sp16, n8);
  }
  // Pass 0b: inputs fp32 -> bf16
  {
    const int n8 = (BATCH * DIM) / 8;
    cvt_bf16_kernel<<<n8 / 256, 256, 0, stream>>>(inputs, (unsigned int*)in16, n8);
  }
  // Pass 1: pack W into B-fragment layout (640 waves, 8 per block)
  {
    const int waves = N_NODES * KSTEPS * NTILES;   // 640
    pack_w_kernel<<<(waves + 7) / 8, 256, 0, stream>>>(W, wp);
  }
  // Pass 2: 16 nodes x 512 row-blocks = 8192 waves -> 1024 blocks
  {
    const int waves = N_NODES * (BATCH / 32);
    snn_gemm_lif_kernel<<<waves / 8, 256, 0, stream>>>(
        in16, sp16, u, i_syn, b, conn, wp, out);
  }
}
